// DFT_747324309649
// MI455X (gfx1250) — matrix operand order repaired and checked by reading, not compiled, not run on hardware
//
#include <hip/hip_runtime.h>

#define BATCH 4096
#define NFFT  2048

typedef __attribute__((ext_vector_type(16))) _Float16 v16h;
typedef __attribute__((ext_vector_type(8)))  _Float16 v8h;
typedef __attribute__((ext_vector_type(4)))  _Float16 v4h;
typedef __attribute__((ext_vector_type(8)))  float    v8f;

union V16U { v16h v; v8h h[2]; };

// ---------------------------------------------------------------------------
// Kernel 0: convert f32 signal (re, im) -> planar f16 in workspace (vectorized)
// ---------------------------------------------------------------------------
__global__ void __launch_bounds__(256)
cvt_f32_to_f16(const float* __restrict__ re, const float* __restrict__ im,
               _Float16* __restrict__ re_h, _Float16* __restrict__ im_h) {
  const int i = (blockIdx.x * 256 + threadIdx.x) * 4;
  float4 r = *(const float4*)(re + i);
  float4 m = *(const float4*)(im + i);
  v4h rh = { (_Float16)r.x, (_Float16)r.y, (_Float16)r.z, (_Float16)r.w };
  v4h mh = { (_Float16)m.x, (_Float16)m.y, (_Float16)m.z, (_Float16)m.w };
  *(v4h*)(re_h + i) = rh;
  *(v4h*)(im_h + i) = mh;
}

// ---------------------------------------------------------------------------
// Kernel 1: build DFT matrix W[m,n] = exp(-2*pi*i*m*n/N) as f16 (re, im).
// Exact integer phase reduction: (m*n) mod N keeps sincos args in (-2pi, 0].
// W is symmetric, which the GEMM exploits to avoid transpose loads.
// ---------------------------------------------------------------------------
__global__ void __launch_bounds__(256)
build_w_f16(_Float16* __restrict__ wr, _Float16* __restrict__ wi) {
  const int idx = blockIdx.x * 256 + threadIdx.x;
  const int m = idx >> 11;          // NFFT = 2048 = 2^11
  const int n = idx & (NFFT - 1);
  const int t = (m * n) & (NFFT - 1);
  const float phase = (float)t * (-6.28318530717958647692f / (float)NFFT);
  float s, c;
  __sincosf(phase, &s, &c);
  wr[idx] = (_Float16)c;
  wi[idx] = (_Float16)s;
}

// ---------------------------------------------------------------------------
// WMMA fragment loaders (direct from global / L2-resident; no LDS staging).
//
// A-matrix 16x32 f16 layout (ISA 7.12.2): lanes 0-15 hold rows M=0..15 with
// K = [k0..k0+7] in VGPR0-3 and K = [k0+16..k0+23] in VGPR4-7; lanes 16-31
// hold the same rows with K shifted by +8. -> two contiguous b128 loads.
// ---------------------------------------------------------------------------
__device__ __forceinline__ v16h loadA(const _Float16* __restrict__ base,
                                      int row0, int k0, int lane) {
  const _Float16* p = base + (size_t)(row0 + (lane & 15)) * NFFT
                           + k0 + ((lane >> 4) << 3);
  V16U u;
  u.h[0] = *(const v8h*)p;          // elements 0..7
  u.h[1] = *(const v8h*)(p + 16);   // elements 8..15
  return u.v;
}

// B-matrix 32x16 f16 layout: lane l holds column N = l%16; same K order as A:
// K = [k0+8h..k0+8h+7] then [k0+16+8h..k0+23+8h], h = lane/16.
// Since W[k][n] == W[n][k] (symmetry), read row n of W contiguously.
__device__ __forceinline__ v16h loadB(const _Float16* __restrict__ base,
                                      int col0, int k0, int lane) {
  const _Float16* p = base + (size_t)(col0 + (lane & 15)) * NFFT
                           + k0 + ((lane >> 4) << 3);
  V16U u;
  u.h[0] = *(const v8h*)p;          // elements 0..7
  u.h[1] = *(const v8h*)(p + 16);   // elements 8..15
  return u.v;
}

__device__ __forceinline__ v8f wmma_f16(v16h a, v16h b, v8f c) {
  return __builtin_amdgcn_wmma_f32_16x16x32_f16(
      /*neg_a=*/false, a, /*neg_b=*/false, b,
      /*c_mod=*/(short)0, c, /*reuse_a=*/false, /*reuse_b=*/false);
}

// ---------------------------------------------------------------------------
// Kernel 2: complex GEMM X = x @ W via v_wmma_f32_16x16x32_f16.
// Block = 256 threads = 8 waves; block tile 128(M) x 64(N); each wave owns a
// 32x32 output tile (2x2 WMMA tiles), K-loop 2048 in steps of 32.
//   re += Ar*Br + (-Ai)*Bi ; im += Ar*Bi + Ai*Br   (2 accumulators/tile)
// ---------------------------------------------------------------------------
__global__ void __launch_bounds__(256)
dft_wmma(const _Float16* __restrict__ xr, const _Float16* __restrict__ xi,
         const _Float16* __restrict__ wr, const _Float16* __restrict__ wi,
         float* __restrict__ outr, float* __restrict__ outi) {
  const int lane = threadIdx.x & 31;
  const int wid  = threadIdx.x >> 5;
  const int rowBase = blockIdx.y * 128 + (wid & 3) * 32;  // batch rows
  const int colBase = blockIdx.x * 64  + (wid >> 2) * 32; // output freq cols

  v8f acc_re[2][2] = {{{}, {}}, {{}, {}}};
  v8f acc_im[2][2] = {{{}, {}}, {{}, {}}};

  for (int k0 = 0; k0 < NFFT; k0 += 32) {
    if (k0 + 32 < NFFT) {  // warm L2 for the next K tile (global_prefetch_b8)
      __builtin_prefetch(xr + (size_t)(rowBase + (lane & 15)) * NFFT + k0 + 32, 0, 3);
      __builtin_prefetch(wr + (size_t)(colBase + (lane & 15)) * NFFT + k0 + 32, 0, 3);
    }

    v16h Ar[2], Ai[2], nAi[2], Br[2], Bi[2];
#pragma unroll
    for (int i = 0; i < 2; ++i) {
      Ar[i]  = loadA(xr, rowBase + i * 16, k0, lane);
      Ai[i]  = loadA(xi, rowBase + i * 16, k0, lane);
      nAi[i] = -Ai[i];   // packed f16 negate; co-executes with XDL WMMAs
    }
#pragma unroll
    for (int j = 0; j < 2; ++j) {
      Br[j] = loadB(wr, colBase + j * 16, k0, lane);
      Bi[j] = loadB(wi, colBase + j * 16, k0, lane);
    }

#pragma unroll
    for (int i = 0; i < 2; ++i) {
#pragma unroll
      for (int j = 0; j < 2; ++j) {
        acc_re[i][j] = wmma_f16(Ar[i],  Br[j], acc_re[i][j]);
        acc_re[i][j] = wmma_f16(nAi[i], Bi[j], acc_re[i][j]);
        acc_im[i][j] = wmma_f16(Ar[i],  Bi[j], acc_im[i][j]);
        acc_im[i][j] = wmma_f16(Ai[i],  Br[j], acc_im[i][j]);
      }
    }
  }

  // Epilogue: C/D layout — VGPR r, lane l -> row = base + (l<16?0:8) + r,
  // col = base + l%16 (ISA 7.12.2 32-bit C/D 16x16 table).
  const int colOff = lane & 15;
  const int rowOff = (lane >> 4) << 3;
#pragma unroll
  for (int i = 0; i < 2; ++i) {
#pragma unroll
    for (int j = 0; j < 2; ++j) {
      const int col = colBase + j * 16 + colOff;
#pragma unroll
      for (int r = 0; r < 8; ++r) {
        const size_t o = (size_t)(rowBase + i * 16 + rowOff + r) * NFFT + col;
        outr[o] = acc_re[i][j][r];
        outi[o] = acc_im[i][j][r];
      }
    }
  }
}

// ---------------------------------------------------------------------------
extern "C" void kernel_launch(void* const* d_in, const int* in_sizes, int n_in,
                              void* d_out, int out_size, void* d_ws, size_t ws_size,
                              hipStream_t stream) {
  const float* sig_re = (const float*)d_in[0];
  const float* sig_im = (const float*)d_in[1];

  // Workspace layout (50.3 MB total):
  //   xr_h, xi_h : [BATCH*NFFT] f16 each (16.8 MB each)
  //   wr_h, wi_h : [NFFT*NFFT]  f16 each ( 8.4 MB each)
  _Float16* xr_h = (_Float16*)d_ws;
  _Float16* xi_h = xr_h + (size_t)BATCH * NFFT;
  _Float16* wr_h = xi_h + (size_t)BATCH * NFFT;
  _Float16* wi_h = wr_h + (size_t)NFFT * NFFT;

  float* outr = (float*)d_out;
  float* outi = outr + (size_t)BATCH * NFFT;

  cvt_f32_to_f16<<<(BATCH * NFFT / 4) / 256, 256, 0, stream>>>(
      sig_re, sig_im, xr_h, xi_h);
  build_w_f16<<<(NFFT * NFFT) / 256, 256, 0, stream>>>(wr_h, wi_h);

  dim3 grid(NFFT / 64, BATCH / 128);  // 32 x 32 blocks
  dft_wmma<<<grid, 256, 0, stream>>>(xr_h, xi_h, wr_h, wi_h, outr, outi);
}
